// KDistanceDetector_13907104105033
// MI455X (gfx1250) — matrix operand order repaired and checked by reading, not compiled, not run on hardware
//
#include <hip/hip_runtime.h>
#include <hip/hip_bf16.h>

// ---------------------------------------------------------------------------
// k-NN distance (k=32): fused Gram-matrix WMMA + per-row 33-smallest select.
// X: [B=16384, D=128] fp32.  out[i] = sqrt(33rd-smallest sq-dist of row i).
//
// fp32 accuracy via split-bf16: x = hi + lo, x.y ~= hi.hi + hi.lo + lo.hi,
// computed with v_wmma_f32_16x16x32_bf16 (K=32 per issue, f32 accumulate).
// ---------------------------------------------------------------------------

typedef __bf16  v8bf  __attribute__((ext_vector_type(8)));
typedef __bf16  v16bf __attribute__((ext_vector_type(16)));
typedef float   v8f   __attribute__((ext_vector_type(8)));

#define DIMS 128
#define NSEL 33              // keep 33 smallest -> max of them is a[:,32]

static __device__ __forceinline__ v16bf cat8(v8bf a, v8bf b) {
    return __builtin_shufflevector(a, b, 0,1,2,3,4,5,6,7,8,9,10,11,12,13,14,15);
}

static __device__ __forceinline__ v8f wmma_bf16(v16bf a, v16bf b, v8f c) {
    // (neg_a, A, neg_b, B, c_mod, C, reuse_a, reuse_b)
    return __builtin_amdgcn_wmma_f32_16x16x32_bf16(false, a, false, b,
                                                   (short)0, c, false, false);
}

// ---------------------------------------------------------------------------
// Kernel 1: per-row squared norms + hi/lo bf16 split.  One 128-thread block
// per row; thread t owns element t.
// ---------------------------------------------------------------------------
__global__ __launch_bounds__(128)
void knn_prep_kernel(const float* __restrict__ X,
                     float* __restrict__ rn,
                     __bf16* __restrict__ Xhi,
                     __bf16* __restrict__ Xlo) {
    const int row = blockIdx.x;
    const int t   = threadIdx.x;
    const int idx = row * DIMS + t;

    float x  = X[idx];
    __bf16 h = (__bf16)x;
    float  r = x - (float)h;
    __bf16 l = (__bf16)r;
    Xhi[idx] = h;
    Xlo[idx] = l;

    // block reduce sum(x*x): intra-wave shuffle, then 4 partials via LDS
    float s = x * x;
    #pragma unroll
    for (int off = 16; off > 0; off >>= 1)
        s += __shfl_xor(s, off, 32);

    __shared__ float red[4];
    if ((t & 31) == 0) red[t >> 5] = s;
    __syncthreads();
    if (t == 0) rn[row] = red[0] + red[1] + red[2] + red[3];
}

// ---------------------------------------------------------------------------
// Kernel 2: main fused GEMM + selection.
// 128 threads = 4 waves per block.  Wave w handles rows
// [ (blockIdx.x*4+w)*16 , +16 ).  Column tiles of 16 are staged into LDS
// once per block (hi+lo), shared by all 4 waves.
// ---------------------------------------------------------------------------
__global__ __launch_bounds__(128)
void knn_main_kernel(const __bf16* __restrict__ Xhi,
                     const __bf16* __restrict__ Xlo,
                     const float*  __restrict__ rn,
                     float* __restrict__ out,
                     int nrows) {
    const int lane = threadIdx.x & 31;
    const int wid  = threadIdx.x >> 5;
    const int h    = lane >> 4;        // lane half (0/1)
    const int ln   = lane & 15;
    const int rowBase = (blockIdx.x * 4 + wid) * 16;

    // Column-tile staging (padded to 136 bf16/row: lane n hits dword banks
    // (n*68)%64 = 4n -> conflict-free b128 reads).
    __shared__ __align__(16) __bf16 sh_hi[16][136];
    __shared__ __align__(16) __bf16 sh_lo[16][136];
    // Per-wave selection state: 33 smallest sq-dists per row.
    // Plain (non-volatile) __shared__ so InferAddressSpaces keeps these in
    // addrspace(3) => ds_load/ds_store.  Re-read semantics are guaranteed by
    // the conditional insertion store (may-alias => reload) and LDS same-wave
    // in-order execution; __syncthreads() fences across tiles.
    __shared__ float best[4][16][NSEL];
    __shared__ float curmax[4][16];
    __shared__ int   maxslot[4][16];

    // init selection state (private to this wave) -- direct indexing only
    #pragma unroll
    for (int r = 0; r < 16; ++r) {
        best[wid][r][lane] = __builtin_inff();
        if (lane == 0) best[wid][r][32] = __builtin_inff();
    }
    if (lane < 16) {
        curmax[wid][lane]  = __builtin_inff();
        maxslot[wid][lane] = 0;
    }

    // ---- load A fragments (16 rows x 128 K, hi+lo) -> registers for good --
    // ISA 16-bit A 16x32 layout: lane M = lane&15; per half, element e maps
    // to K = e + (e>=8 ? 8 : 0) + 8*h  => two contiguous 8-element loads.
    v16bf Ahi[4], Alo[4];
    {
        const int arow = rowBase + ln;
        #pragma unroll
        for (int c = 0; c < 4; ++c) {
            const int k0 = 32 * c;
            const __bf16* ph = Xhi + (size_t)arow * DIMS + k0 + 8 * h;
            const __bf16* pl = Xlo + (size_t)arow * DIMS + k0 + 8 * h;
            v8bf h0 = *(const v8bf*)(ph);
            v8bf h1 = *(const v8bf*)(ph + 16);
            v8bf l0 = *(const v8bf*)(pl);
            v8bf l1 = *(const v8bf*)(pl + 16);
            Ahi[c] = cat8(h0, h1);
            Alo[c] = cat8(l0, l1);
        }
    }

    // per-lane row norms for the 8 accumulator VGPRs (row = v + 8*h)
    float rnrow[8];
    #pragma unroll
    for (int v = 0; v < 8; ++v)
        rnrow[v] = rn[rowBase + v + 8 * h];

    const int ntiles = nrows / 16;
    for (int jt = 0; jt < ntiles; ++jt) {
        const int colBase = jt * 16;

        __syncthreads();  // previous tile fully consumed before overwrite
        // stage 16 cols x 128 K (hi+lo) cooperatively: 256 b128 chunks each
        #pragma unroll
        for (int it = 0; it < 2; ++it) {
            int idx = threadIdx.x + 128 * it;     // 0..255
            int r   = idx >> 4;
            int ch  = idx & 15;
            const size_t g = (size_t)(colBase + r) * DIMS + ch * 8;
            *(v8bf*)(&sh_hi[r][ch * 8]) = *(const v8bf*)(Xhi + g);
            *(v8bf*)(&sh_lo[r][ch * 8]) = *(const v8bf*)(Xlo + g);
        }
        __syncthreads();

        // ---- 12 WMMAs: hi.hi + hi.lo + lo.hi over 4 K-chunks -------------
        // ISA 16-bit B 32x16 layout: lane N = lane&15; K = e + (e>=8 ? 8 : 0) + 8*h.
        v8f acc = {0.f, 0.f, 0.f, 0.f, 0.f, 0.f, 0.f, 0.f};
        #pragma unroll
        for (int c = 0; c < 4; ++c) {
            const int k0 = 32 * c + 8 * h;
            v16bf bhi = cat8(*(const v8bf*)(&sh_hi[ln][k0]),
                             *(const v8bf*)(&sh_hi[ln][k0 + 16]));
            v16bf blo = cat8(*(const v8bf*)(&sh_lo[ln][k0]),
                             *(const v8bf*)(&sh_lo[ln][k0 + 16]));
            acc = wmma_bf16(Ahi[c], bhi, acc);
            acc = wmma_bf16(Ahi[c], blo, acc);
            acc = wmma_bf16(Alo[c], bhi, acc);
        }

        const float rncol = rn[colBase + ln];

        // ---- fused selection: sq dists never leave the wave --------------
        #pragma unroll
        for (int v = 0; v < 8; ++v) {
            float sq = rnrow[v] + rncol - 2.0f * acc[v];
            sq = fmaxf(sq, 0.0f);
            if (colBase + ln == rowBase + v + 8 * h) sq = __builtin_inff();

            float cm = curmax[wid][v + 8 * h];              // ds_load
            unsigned mask = (unsigned)__builtin_amdgcn_ballot_w32(sq < cm);
            while (mask) {                              // rare path
                int src = __builtin_ctz(mask);
                mask &= mask - 1;
                float c = __builtin_bit_cast(
                    float, __builtin_amdgcn_readlane(
                               __builtin_bit_cast(int, sq), src));
                int rloc = v + 8 * (src >> 4);
                if (c < curmax[wid][rloc]) {            // uniform re-check
                    if (lane == 0) {                    // serialize insert
                        int slot = maxslot[wid][rloc];
                        best[wid][rloc][slot] = c;
                        float m = -__builtin_inff();
                        int ms = 0;
                        for (int t = 0; t < NSEL; ++t) {
                            float bv = best[wid][rloc][t];
                            if (bv > m) { m = bv; ms = t; }
                        }
                        curmax[wid][rloc]  = m;
                        maxslot[wid][rloc] = ms;
                    }
                }
            }
        }
    }

    // max of the 33 smallest sq-dists == 33rd smallest == a[:,32]
    if (lane < 16)
        out[rowBase + lane] = sqrtf(curmax[wid][lane]);
}

// ---------------------------------------------------------------------------
extern "C" void kernel_launch(void* const* d_in, const int* in_sizes, int n_in,
                              void* d_out, int out_size, void* d_ws, size_t ws_size,
                              hipStream_t stream) {
    const float* X = (const float*)d_in[0];
    float* out     = (float*)d_out;
    const int nrows = in_sizes[0] / DIMS;   // 16384

    // workspace layout: rn | Xhi | Xlo
    float*  rn  = (float*)d_ws;
    __bf16* Xhi = (__bf16*)((char*)d_ws + 65536);
    __bf16* Xlo = Xhi + (size_t)nrows * DIMS;

    knn_prep_kernel<<<nrows, 128, 0, stream>>>(X, rn, Xhi, Xlo);
    knn_main_kernel<<<nrows / 64, 128, 0, stream>>>(Xhi, Xlo, rn, out, nrows);
}
